// ScaledDotProductAttention_81363860455625
// MI455X (gfx1250) — matrix operand order repaired and checked by reading, not compiled, not run on hardware
//
#include <hip/hip_runtime.h>
#include <hip/hip_bf16.h>

// CDNA5 (gfx1250) fused causal attention producing (output, attn).
// wave32, WMMA bf16 16x16x32 with f32 accumulate, causal work-skipping,
// non-temporal streaming of the 1GB attn output.

typedef __attribute__((ext_vector_type(16))) __bf16 v16bf;
typedef __attribute__((ext_vector_type(8)))  float  v8f;

#define N_SEQ   2048
#define DHEAD   64
#define SROW    2064            // padded LDS row stride (floats); 2064 % 64 == 16
#define NEG_INF -1000000000.0f

static __device__ __forceinline__ __bf16 f2bf(float x) { return (__bf16)x; }

__global__ __launch_bounds__(256) void attn_fwd_gfx1250(
    const float* __restrict__ Q, const float* __restrict__ K,
    const float* __restrict__ V, float* __restrict__ Out,
    float* __restrict__ Attn) {
  extern __shared__ float smem[];
  float* Srow = smem;                   // 16 x SROW scores -> probs
  float* red  = smem + 16 * SROW;       // 4 x 256 floats PV fold scratch
  float* rinv = red + 4 * 256;          // 16 floats: per-row 1/sum

  const int tid  = threadIdx.x;
  const int lane = tid & 31;            // wave32
  const int wave = tid >> 5;            // 0..7
  const int bh   = blockIdx.x >> 7;     // b*H + h
  const int qt   = blockIdx.x & 127;    // 16-row query tile

  const size_t head_base = (size_t)bh * N_SEQ * DHEAD;
  const int mrow = lane & 15;           // A row or B/C col within 16-tile
  const int hi   = (lane >= 16);

  // ---------------- Phase 1: S = (Q/8) @ K^T, lower-triangular only -------
  v16bf aq[2];
  {
    // A(16x32) layout: lane half selects +8 in K; vgpr pairs cover 0..7,16..23
    const float* qp = Q + head_base + (size_t)(qt * 16 + mrow) * DHEAD + (hi ? 8 : 0);
#pragma unroll
    for (int c = 0; c < 2; ++c) {
      const float4 q0 = *(const float4*)(qp + c * 32);
      const float4 q1 = *(const float4*)(qp + c * 32 + 4);
      const float4 q2 = *(const float4*)(qp + c * 32 + 16);
      const float4 q3 = *(const float4*)(qp + c * 32 + 20);
      aq[c][0]  = f2bf(q0.x * 0.125f); aq[c][1]  = f2bf(q0.y * 0.125f);
      aq[c][2]  = f2bf(q0.z * 0.125f); aq[c][3]  = f2bf(q0.w * 0.125f);
      aq[c][4]  = f2bf(q1.x * 0.125f); aq[c][5]  = f2bf(q1.y * 0.125f);
      aq[c][6]  = f2bf(q1.z * 0.125f); aq[c][7]  = f2bf(q1.w * 0.125f);
      aq[c][8]  = f2bf(q2.x * 0.125f); aq[c][9]  = f2bf(q2.y * 0.125f);
      aq[c][10] = f2bf(q2.z * 0.125f); aq[c][11] = f2bf(q2.w * 0.125f);
      aq[c][12] = f2bf(q3.x * 0.125f); aq[c][13] = f2bf(q3.y * 0.125f);
      aq[c][14] = f2bf(q3.z * 0.125f); aq[c][15] = f2bf(q3.w * 0.125f);
    }
  }

  for (int kt = wave; kt <= qt; kt += 8) {          // wave-uniform causal skip
    // B(32x16) layout: lane half selects +8 in K; vgpr pairs cover 0..7,16..23
    const float* kp = K + head_base + (size_t)(kt * 16 + mrow) * DHEAD + (hi ? 8 : 0);
    v8f acc = {};
#pragma unroll
    for (int c = 0; c < 2; ++c) {
      const float4 b0 = *(const float4*)(kp + c * 32);
      const float4 b1 = *(const float4*)(kp + c * 32 + 4);
      const float4 b2 = *(const float4*)(kp + c * 32 + 16);
      const float4 b3 = *(const float4*)(kp + c * 32 + 20);
      v16bf bk;
      bk[0]  = f2bf(b0.x); bk[1]  = f2bf(b0.y); bk[2]  = f2bf(b0.z); bk[3]  = f2bf(b0.w);
      bk[4]  = f2bf(b1.x); bk[5]  = f2bf(b1.y); bk[6]  = f2bf(b1.z); bk[7]  = f2bf(b1.w);
      bk[8]  = f2bf(b2.x); bk[9]  = f2bf(b2.y); bk[10] = f2bf(b2.z); bk[11] = f2bf(b2.w);
      bk[12] = f2bf(b3.x); bk[13] = f2bf(b3.y); bk[14] = f2bf(b3.z); bk[15] = f2bf(b3.w);
      acc = __builtin_amdgcn_wmma_f32_16x16x32_bf16(false, aq[c], false, bk,
                                                    (short)0, acc, false, false);
    }
    if (kt < qt) {                                  // strictly below diagonal
#pragma unroll
      for (int v = 0; v < 8; ++v) {
        const int M = v + (hi ? 8 : 0);             // C layout: M = v (+8)
        Srow[M * SROW + kt * 16 + mrow] = acc[v];
      }
    } else {                                        // diagonal tile: mask
#pragma unroll
      for (int v = 0; v < 8; ++v) {
        const int M = v + (hi ? 8 : 0);
        Srow[M * SROW + kt * 16 + mrow] = (mrow <= M) ? acc[v] : NEG_INF;
      }
    }
  }
  __syncthreads();

  // ---------------- Phase 2a: per-row max / exp-sum (16 threads per row) --
  {
    const int row = tid >> 4;
    const int sub = tid & 15;
    float* sr = Srow + row * SROW;

    float m = NEG_INF;
    for (int j = 0; j <= qt; ++j) m = fmaxf(m, sr[sub + j * 16]);
#pragma unroll
    for (int o = 8; o >= 1; o >>= 1) m = fmaxf(m, __shfl_xor(m, o, 32));

    float ssum = 0.0f;
    for (int j = 0; j <= qt; ++j) {
      const int c = sub + j * 16;
      const float e = __expf(sr[c] - m);            // masked -> 0 exactly
      sr[c] = e;
      ssum += e;
    }
#pragma unroll
    for (int o = 8; o >= 1; o >>= 1) ssum += __shfl_xor(ssum, o, 32);
    if (sub == 0) rinv[row] = 1.0f / ssum;
  }
  __syncthreads();

  // ---------------- Phase 2b: normalize + stream attn (128B/wave stores) --
  const int gmax  = (qt * 16 + 15) >> 5;            // last live 32-col group
  {
    const int jlim = gmax + 1;
#pragma unroll
    for (int rr = 0; rr < 2; ++rr) {
      const int row = wave + rr * 8;                // wave owns rows w, w+8
      const int qg  = qt * 16 + row;
      const float inv = rinv[row];
      float* sr   = Srow + row * SROW;
      float* arow = Attn + ((size_t)bh * N_SEQ + (size_t)qg) * N_SEQ;
      for (int j = 0; j < jlim; ++j) {              // live region
        const int c = lane + j * 32;
        const float p = (c <= qg) ? sr[c] * inv : 0.0f;
        sr[c] = p;                                  // zero boundary for PV
        __builtin_nontemporal_store(p, arow + c);
      }
      for (int j = jlim; j < N_SEQ / 32; ++j)       // masked region: zeros
        __builtin_nontemporal_store(0.0f, arow + lane + j * 32);
    }
  }
  __syncthreads();

  // ---------------- Phase 3: O = P @ V (causal-capped K sweep) ------------
  const int ct = wave & 3;              // output d-tile: cols ct*16..+15
  const int kh = wave >> 2;             // K half owner
  v8f oacc = {};
  {
    const float* pp = Srow + mrow * SROW + (hi ? 8 : 0);
    for (int kc = 0; kc < 32; ++kc) {
      const int g = kh * 32 + kc;       // 32-key chunk index
      if (g > gmax) break;              // wave-uniform causal cap
      const int kbase = g * 32;
      const float4 p0 = *(const float4*)(pp + kbase);
      const float4 p1 = *(const float4*)(pp + kbase + 4);
      const float4 p2 = *(const float4*)(pp + kbase + 16);
      const float4 p3 = *(const float4*)(pp + kbase + 20);
      v16bf ap;
      ap[0]  = f2bf(p0.x); ap[1]  = f2bf(p0.y); ap[2]  = f2bf(p0.z); ap[3]  = f2bf(p0.w);
      ap[4]  = f2bf(p1.x); ap[5]  = f2bf(p1.y); ap[6]  = f2bf(p1.z); ap[7]  = f2bf(p1.w);
      ap[8]  = f2bf(p2.x); ap[9]  = f2bf(p2.y); ap[10] = f2bf(p2.z); ap[11] = f2bf(p2.w);
      ap[12] = f2bf(p3.x); ap[13] = f2bf(p3.y); ap[14] = f2bf(p3.z); ap[15] = f2bf(p3.w);
      v16bf bv;
#pragma unroll
      for (int v = 0; v < 8; ++v) {
        const int kb = (hi ? 8 : 0) + v * 2 + (v >= 4 ? 8 : 0);       // B layout K offset
        const float* vrow = V + head_base + (size_t)(kbase + kb) * DHEAD + ct * 16 + mrow;
        bv[2 * v]     = f2bf(vrow[0]);
        bv[2 * v + 1] = f2bf(vrow[DHEAD]);
      }
      oacc = __builtin_amdgcn_wmma_f32_16x16x32_bf16(false, ap, false, bv,
                                                     (short)0, oacc, false, false);
    }
  }

  if (kh == 1) {                        // fold the two K-halves via LDS
    float* dst = red + ct * 256 + lane * 8;
#pragma unroll
    for (int v = 0; v < 8; ++v) dst[v] = oacc[v];
  }
  __syncthreads();
  if (kh == 0) {
    const float* src = red + ct * 256 + lane * 8;
#pragma unroll
    for (int v = 0; v < 8; ++v) oacc[v] += src[v];
#pragma unroll
    for (int v = 0; v < 8; ++v) {
      const int M = v + (hi ? 8 : 0);
      __builtin_nontemporal_store(
          oacc[v], Out + head_base + (size_t)(qt * 16 + M) * DHEAD + ct * 16 + mrow);
    }
  }
}

extern "C" void kernel_launch(void* const* d_in, const int* in_sizes, int n_in,
                              void* d_out, int out_size, void* d_ws, size_t ws_size,
                              hipStream_t stream) {
  (void)in_sizes; (void)n_in; (void)out_size; (void)d_ws; (void)ws_size;
  const float* q = (const float*)d_in[0];
  const float* k = (const float*)d_in[1];
  const float* v = (const float*)d_in[2];
  // d_in[3] is the causal mask; causality is applied analytically in-kernel.

  float* out  = (float*)d_out;                       // [B,H,N,D] first
  float* attn = out + (size_t)4 * 16 * 2048 * 64;    // then [B,H,N,N]

  const size_t lds_bytes = (size_t)(16 * SROW + 4 * 256 + 16) * sizeof(float);
  (void)hipFuncSetAttribute((const void*)attn_fwd_gfx1250,
                            hipFuncAttributeMaxDynamicSharedMemorySize,
                            (int)lds_bytes);

  attn_fwd_gfx1250<<<dim3(64 * 128), dim3(256), lds_bytes, stream>>>(q, k, v, out, attn);
}
